// ByteEncoder_15728170238200
// MI455X (gfx1250) — matrix operand order repaired and checked by reading, not compiled, not run on hardware
//
#include <hip/hip_runtime.h>
#include <hip/hip_bf16.h>

typedef _Float16 f16;
typedef __attribute__((ext_vector_type(16))) _Float16 v16h;
typedef __attribute__((ext_vector_type(8)))  _Float16 v8h;
typedef __attribute__((ext_vector_type(8)))  float    v8f;

#define S_LEN   16384
#define D_MODEL 512
#define VOCAB   256
#define PLMAX   16
#define PMINP   4
#define PTHR    0.5f
#define NHEAD   8
#define HDIM    64
#define FDIM    2048
#define NPATCH  4096            /* S/MINP upper bound */
#define NTOK    (NPATCH*PLMAX)  /* 65536 padded token slots */
#define INV_LOG_V 0.18033688f   /* 1/ln(256) */

// ---------------------------------------------------------------------------
// Generic f16 WMMA GEMM:  C[M,N] = act(A[M,K] @ B[N,K]^T + bias) (+res), f32 acc
// One wave computes a 32x64 output tile (2 m-tiles x 4 n-tiles): each loaded
// B fragment feeds 2 WMMAs, each A fragment feeds 4. K is a multiple of 64;
// the K loop is software-pipelined with ping-pong fragment buffers so the 12
// loads for chunk k+32 overlap the 8 WMMAs of chunk k (partial loadcnt waits).
// ---------------------------------------------------------------------------
template<int ACT, bool BIAS, bool RES, bool ST16, bool ST32>
__global__ void __launch_bounds__(256)
gemm_f16_kernel(const f16* __restrict__ A, const f16* __restrict__ B,
                const float* __restrict__ bias, const float* __restrict__ res,
                f16* __restrict__ C16, float* __restrict__ C32,
                int M, int N, int K) {
  const int lane = threadIdx.x & 31;
  const int wave = threadIdx.x >> 5;
  const int tiles_n = N >> 6;                 // 64-wide wave tiles
  const int wid = blockIdx.x * 8 + wave;
  const int tm  = wid / tiles_n;              // 32-row wave tiles
  const int tn  = wid - tm * tiles_n;
  if (tm * 32 >= M) return;
  const int half = lane >> 4;                 // 0 or 1 (half-wave)
  const int lnib = lane & 15;
  const long arow0 = (long)(tm * 32 + lnib) * K + half * 8;
  const long arow1 = arow0 + (long)16 * K;
  long brow[4];
#pragma unroll
  for (int t = 0; t < 4; ++t)
    brow[t] = (long)(tn * 64 + t * 16 + lnib) * K + half * 8;

  v8f acc[2][4];
#pragma unroll
  for (int i = 0; i < 2; ++i)
#pragma unroll
    for (int t = 0; t < 4; ++t)
#pragma unroll
      for (int r = 0; r < 8; ++r) acc[i][t][r] = 0.0f;

  union AV { v16h v; v8h h[2]; };
  // A 16x32 f16 layout: lane holds row m=lane%16; K chunks [8*half,+8) and [16+8*half,+8)
  auto loadA = [&](AV* a, int k0) {
    const f16* ap0 = A + arow0 + k0;
    a[0].h[0] = *(const v8h*)(ap0);
    a[0].h[1] = *(const v8h*)(ap0 + 16);
    const f16* ap1 = A + arow1 + k0;
    a[1].h[0] = *(const v8h*)(ap1);
    a[1].h[1] = *(const v8h*)(ap1 + 16);
  };
  // B 32x16 layout: lane holds col n=lane%16; K chunks [8*half,+8) and [16+8*half,+8)
  auto loadB = [&](AV* b, int k0) {
#pragma unroll
    for (int t = 0; t < 4; ++t) {
      const f16* bp = B + brow[t] + k0;
      b[t].h[0] = *(const v8h*)(bp);
      b[t].h[1] = *(const v8h*)(bp + 16);
    }
  };
  auto mma = [&](AV* a, AV* b) {
#pragma unroll
    for (int i = 0; i < 2; ++i)
#pragma unroll
      for (int t = 0; t < 4; ++t)
        acc[i][t] = __builtin_amdgcn_wmma_f32_16x16x32_f16(
            false, a[i].v, false, b[t].v, (short)0, acc[i][t], false, false);
  };

  AV a0[2], a1[2], b0[4], b1[4];
  loadA(a0, 0); loadB(b0, 0);
  for (int k0 = 0; k0 < K; k0 += 64) {
    if (k0 + 128 <= K) {                       // WGP-scope prefetch 2 chunks ahead
      __builtin_prefetch(A + arow0 + k0 + 128, 0, 3);
      __builtin_prefetch(A + arow1 + k0 + 128, 0, 3);
      __builtin_prefetch(B + brow[0] + k0 + 128, 0, 3);
      __builtin_prefetch(B + brow[2] + k0 + 128, 0, 3);
    }
    loadA(a1, k0 + 32); loadB(b1, k0 + 32);    // overlap with WMMAs below
    mma(a0, b0);
    if (k0 + 64 < K) { loadA(a0, k0 + 64); loadB(b0, k0 + 64); }
    mma(a1, b1);
  }

  // C/D layout: lane n=lane%16, vgpr r -> m = r + 8*half
#pragma unroll
  for (int i = 0; i < 2; ++i) {
    const int mbase = tm * 32 + i * 16 + half * 8;
#pragma unroll
    for (int t = 0; t < 4; ++t) {
      const int n = tn * 64 + t * 16 + lnib;
      const float bv = BIAS ? bias[n] : 0.0f;
#pragma unroll
      for (int r = 0; r < 8; ++r) {
        const long idx = (long)(mbase + r) * N + n;
        float x = acc[i][t][r] + bv;
        if (ACT == 1) x = fmaxf(x, 0.0f);                              // ReLU
        if (ACT == 2) x = 0.5f * x * (1.0f + erff(x * 0.70710678f));   // exact GELU
        if (RES) x += res[idx];
        if (ST32) C32[idx] = x;
        if (ST16) C16[idx] = (f16)x;
      }
    }
  }
}

// ---------------------------------------------------------------------------
// Small helper kernels
// ---------------------------------------------------------------------------
__global__ void cvt_f16_kernel(const float* __restrict__ s, f16* __restrict__ d, int n) {
  int i = blockIdx.x * 256 + threadIdx.x;
  if (i < n) d[i] = (f16)s[i];
}

__global__ void zero_i32_kernel(int* __restrict__ p, int n) {
  int i = blockIdx.x * 256 + threadIdx.x;
  if (i < n) p[i] = 0;
}

// gather byte embeddings (f16) for the entropy model input
__global__ void gather_bytes_kernel(const int* __restrict__ seq,
                                    const f16* __restrict__ emb16,
                                    f16* __restrict__ xb16) {
  long i = (long)blockIdx.x * 256 + threadIdx.x;   // over S*D
  int s = (int)(i >> 9), d = (int)(i & 511);
  xb16[i] = emb16[(long)seq[s] * D_MODEL + d];
}

// per-byte entropy from logits [S, 256]; one wave per row
__global__ void entropy_kernel(const float* __restrict__ logits, float* __restrict__ ent) {
  int s = blockIdx.x * 8 + (threadIdx.x >> 5);
  int lane = threadIdx.x & 31;
  const float* lr = logits + (long)s * VOCAB;
  float l[8];
  float mx = -3.4e38f;
#pragma unroll
  for (int j = 0; j < 8; ++j) { l[j] = lr[j * 32 + lane]; mx = fmaxf(mx, l[j]); }
#pragma unroll
  for (int o = 16; o > 0; o >>= 1) mx = fmaxf(mx, __shfl_xor(mx, o, 32));
  float se = 0.0f;
#pragma unroll
  for (int j = 0; j < 8; ++j) se += __expf(l[j] - mx);
#pragma unroll
  for (int o = 16; o > 0; o >>= 1) se += __shfl_xor(se, o, 32);
  float logZ = mx + logf(se);
  float sl = 0.0f;
#pragma unroll
  for (int j = 0; j < 8; ++j) sl += __expf(l[j] - logZ) * l[j];
#pragma unroll
  for (int o = 16; o > 0; o >>= 1) sl += __shfl_xor(sl, o, 32);
  if (lane == 0) ent[s] = (logZ - sl) * INV_LOG_V;
}

// strictly sequential entropy-gated segmentation (matches the lax.scan).
// Loads are batched 4-ahead so the serial state machine isn't one dependent
// memory round trip per byte.
__global__ void scan_kernel(const float* __restrict__ ent, const int* __restrict__ seq,
                            int* __restrict__ tokens, int* __restrict__ lengths) {
  if (threadIdx.x != 0 || blockIdx.x != 0) return;
  int c = 0, nb = 0;
  for (int s = 0; s < S_LEN; s += 4) {
    float e[4]; int t[4];
#pragma unroll
    for (int j = 0; j < 4; ++j) { e[j] = ent[s + j]; t[j] = seq[s + j]; }
#pragma unroll
    for (int j = 0; j < 4; ++j) {
      int c1 = c + 1;
      bool b = ((c1 >= PMINP) && (e[j] < PTHR)) || (c1 >= PLMAX);
      tokens[nb * PLMAX + c] = t[j];
      lengths[nb] = c1;
      if (b) { nb++; c = 0; } else c = c1;
    }
  }
}

// dense padded patch embeddings (f32 + f16)
__global__ void embed_patches_kernel(const int* __restrict__ tokens,
                                     const float* __restrict__ emb,
                                     float* __restrict__ x32, f16* __restrict__ x16) {
  long i = (long)blockIdx.x * 256 + threadIdx.x;   // over NTOK*D
  int t = (int)(i >> 9), d = (int)(i & 511);
  float v = emb[(long)tokens[t] * D_MODEL + d];
  x32[i] = v;
  x16[i] = (f16)v;
}

// masked attention: one block per patch, one wave per head, lanes split head-dim
__global__ void __launch_bounds__(256)
attention_kernel(const float* __restrict__ qkv, const int* __restrict__ lengths,
                 f16* __restrict__ att16) {
  const int p = blockIdx.x;
  const int h = threadIdx.x >> 5;
  const int lane = threadIdx.x & 31;
  const int l  = lane & 15;          // query row
  const int dh = (lane >> 4) * 32;   // half of the 64-dim head
  const int len = lengths[p];
  const long base = (long)p * PLMAX * (3 * D_MODEL);

  float q[32];
  const float* qp = qkv + base + (long)l * (3 * D_MODEL) + h * HDIM + dh;
#pragma unroll
  for (int j = 0; j < 32; ++j) q[j] = qp[j];

  float sc[16];
#pragma unroll
  for (int m = 0; m < 16; ++m) {
    const float* kp = qkv + base + (long)m * (3 * D_MODEL) + D_MODEL + h * HDIM + dh;
    float s = 0.0f;
#pragma unroll
    for (int j = 0; j < 32; ++j) s += q[j] * kp[j];
    sc[m] = s;
  }
#pragma unroll
  for (int m = 0; m < 16; ++m) sc[m] += __shfl_xor(sc[m], 16, 32);  // combine d-halves

  float mx = -3.4e38f;
#pragma unroll
  for (int m = 0; m < 16; ++m) {
    sc[m] = (m < len) ? sc[m] * 0.125f : -1e9f;   // /sqrt(64), key mask
    mx = fmaxf(mx, sc[m]);
  }
  float se = 0.0f;
#pragma unroll
  for (int m = 0; m < 16; ++m) { sc[m] = __expf(sc[m] - mx); se += sc[m]; }
  const float inv = 1.0f / se;

  float o[32];
#pragma unroll
  for (int j = 0; j < 32; ++j) o[j] = 0.0f;
#pragma unroll
  for (int m = 0; m < 16; ++m) {
    float a = sc[m] * inv;
    const float* vp = qkv + base + (long)m * (3 * D_MODEL) + 2 * D_MODEL + h * HDIM + dh;
#pragma unroll
    for (int j = 0; j < 32; ++j) o[j] += a * vp[j];
  }
  f16* op = att16 + (long)(p * PLMAX + l) * D_MODEL + h * HDIM + dh;
#pragma unroll
  for (int j = 0; j < 32; ++j) op[j] = (f16)o[j];
}

// LayerNorm over D=512; one wave per token row; writes f32 + f16
__global__ void __launch_bounds__(256)
ln_kernel(const float* __restrict__ y, const float* __restrict__ g,
          const float* __restrict__ b, float* __restrict__ x32,
          f16* __restrict__ x16, int M) {
  int row = blockIdx.x * 8 + (threadIdx.x >> 5);
  int lane = threadIdx.x & 31;
  if (row >= M) return;
  const float* yr = y + (long)row * D_MODEL;
  float v[16];
  float s = 0.0f;
#pragma unroll
  for (int j = 0; j < 16; ++j) { v[j] = yr[j * 32 + lane]; s += v[j]; }
#pragma unroll
  for (int o = 16; o > 0; o >>= 1) s += __shfl_xor(s, o, 32);
  float mean = s * (1.0f / D_MODEL);
  float vs = 0.0f;
#pragma unroll
  for (int j = 0; j < 16; ++j) { float d = v[j] - mean; vs += d * d; }
#pragma unroll
  for (int o = 16; o > 0; o >>= 1) vs += __shfl_xor(vs, o, 32);
  float inv = rsqrtf(vs * (1.0f / D_MODEL) + 1e-5f);
#pragma unroll
  for (int j = 0; j < 16; ++j) {
    int d = j * 32 + lane;
    float ov = (v[j] - mean) * inv * g[d] + b[d];
    x32[(long)row * D_MODEL + d] = ov;
    x16[(long)row * D_MODEL + d] = (f16)ov;
  }
}

// masked mean pool + outputs; one wave per patch
__global__ void __launch_bounds__(256)
pool_kernel(const float* __restrict__ x32, const int* __restrict__ lengths,
            float* __restrict__ emb_out, int* __restrict__ len_out) {
  int p = blockIdx.x * 8 + (threadIdx.x >> 5);
  int lane = threadIdx.x & 31;
  if (p >= NPATCH) return;
  int len = lengths[p];
  float acc[16];
#pragma unroll
  for (int j = 0; j < 16; ++j) acc[j] = 0.0f;
  for (int l = 0; l < len; ++l) {
    const float* xr = x32 + (long)(p * PLMAX + l) * D_MODEL;
#pragma unroll
    for (int j = 0; j < 16; ++j) acc[j] += xr[j * 32 + lane];
  }
  float invc = (len > 0) ? 1.0f / (float)len : 0.0f;
#pragma unroll
  for (int j = 0; j < 16; ++j)
    emb_out[(long)p * D_MODEL + j * 32 + lane] = acc[j] * invc;
  if (lane == 0) len_out[p] = len;
}

// ---------------------------------------------------------------------------
extern "C" void kernel_launch(void* const* d_in, const int* in_sizes, int n_in,
                              void* d_out, int out_size, void* d_ws, size_t ws_size,
                              hipStream_t stream) {
  (void)in_sizes; (void)n_in; (void)out_size; (void)ws_size;
  const int*   seq      = (const int*)  d_in[0];
  const float* byte_emb = (const float*)d_in[1];
  const float* ent_w1   = (const float*)d_in[2];
  const float* ent_b1   = (const float*)d_in[3];
  const float* ent_w2   = (const float*)d_in[4];
  const float* ent_b2   = (const float*)d_in[5];
  const float* wqkv     = (const float*)d_in[6];
  const float* bqkv     = (const float*)d_in[7];
  const float* wo       = (const float*)d_in[8];
  const float* bo       = (const float*)d_in[9];
  const float* ln1g     = (const float*)d_in[10];
  const float* ln1b     = (const float*)d_in[11];
  const float* w1       = (const float*)d_in[12];
  const float* b1       = (const float*)d_in[13];
  const float* w2       = (const float*)d_in[14];
  const float* b2       = (const float*)d_in[15];
  const float* ln2g     = (const float*)d_in[16];
  const float* ln2b     = (const float*)d_in[17];

  char* wsb = (char*)d_ws;
  size_t off = 0;
  auto alloc = [&](size_t bytes) -> char* {
    char* p = wsb + off;
    off += (bytes + 255) & ~(size_t)255;
    return p;
  };
  // f16 weights
  f16* emb16   = (f16*)alloc((size_t)VOCAB * D_MODEL * 2);
  f16* entw1_h = (f16*)alloc((size_t)2 * D_MODEL * D_MODEL * 2);        // 1024x512
  f16* entw2_h = (f16*)alloc((size_t)VOCAB * 2 * D_MODEL * 2);          // 256x1024
  f16* wqkv_h  = (f16*)alloc((size_t)2 * 3 * D_MODEL * D_MODEL * 2);
  f16* wo_h    = (f16*)alloc((size_t)2 * D_MODEL * D_MODEL * 2);
  f16* w1_h    = (f16*)alloc((size_t)2 * FDIM * D_MODEL * 2);
  f16* w2_h    = (f16*)alloc((size_t)2 * D_MODEL * FDIM * 2);
  // small buffers
  float* ent     = (float*)alloc((size_t)S_LEN * 4);
  int*   tokens  = (int*)  alloc((size_t)NTOK * 4);
  int*   lengths = (int*)  alloc((size_t)NPATCH * 4);
  // activations
  float* x32   = (float*)alloc((size_t)NTOK * D_MODEL * 4);
  f16*   x16   = (f16*)  alloc((size_t)NTOK * D_MODEL * 2);
  f16*   att16 = (f16*)  alloc((size_t)NTOK * D_MODEL * 2);
  // big region (402 MB), multiply-aliased:
  //   phase 1 (entropy): xb16 | h16 | logits
  //   per-layer: qkv32 (whole) -> then y32 (first 134MB) + hf16 (next 268MB)
  char* big = alloc((size_t)NTOK * 3 * D_MODEL * 4);
  f16*   xb16   = (f16*)big;                                            // 16 MB
  f16*   h16    = (f16*)(big + 17 * 1024 * 1024);                       // 33.6 MB
  float* logits = (float*)(big + 64 * 1024 * 1024);                     // 16.8 MB
  float* qkv32  = (float*)big;
  float* y32    = (float*)big;
  f16*   hf16   = (f16*)(big + (size_t)NTOK * D_MODEL * 4);

  auto gblk = [](int M, int N) { return (M / 32) * (N / 64) / 8; };

  // ---- weight conversion (f32 -> f16) ----
  cvt_f16_kernel<<<(VOCAB * D_MODEL) / 256, 256, 0, stream>>>(byte_emb, emb16, VOCAB * D_MODEL);
  cvt_f16_kernel<<<(2 * D_MODEL * D_MODEL) / 256, 256, 0, stream>>>(ent_w1, entw1_h, 2 * D_MODEL * D_MODEL);
  cvt_f16_kernel<<<(VOCAB * 2 * D_MODEL) / 256, 256, 0, stream>>>(ent_w2, entw2_h, VOCAB * 2 * D_MODEL);
  cvt_f16_kernel<<<(2 * 3 * D_MODEL * D_MODEL) / 256, 256, 0, stream>>>(wqkv, wqkv_h, 2 * 3 * D_MODEL * D_MODEL);
  cvt_f16_kernel<<<(2 * D_MODEL * D_MODEL) / 256, 256, 0, stream>>>(wo, wo_h, 2 * D_MODEL * D_MODEL);
  cvt_f16_kernel<<<(2 * FDIM * D_MODEL) / 256, 256, 0, stream>>>(w1, w1_h, 2 * FDIM * D_MODEL);
  cvt_f16_kernel<<<(2 * D_MODEL * FDIM) / 256, 256, 0, stream>>>(w2, w2_h, 2 * D_MODEL * FDIM);

  // ---- entropy model ----
  gather_bytes_kernel<<<(S_LEN * D_MODEL) / 256, 256, 0, stream>>>(seq, emb16, xb16);
  gemm_f16_kernel<2, true, false, true, false><<<gblk(S_LEN, 2 * D_MODEL), 256, 0, stream>>>(
      xb16, entw1_h, ent_b1, nullptr, h16, nullptr, S_LEN, 2 * D_MODEL, D_MODEL);
  gemm_f16_kernel<0, true, false, false, true><<<gblk(S_LEN, VOCAB), 256, 0, stream>>>(
      h16, entw2_h, ent_b2, nullptr, nullptr, logits, S_LEN, VOCAB, 2 * D_MODEL);
  entropy_kernel<<<S_LEN / 8, 256, 0, stream>>>(logits, ent);

  // ---- segmentation + patch embed ----
  zero_i32_kernel<<<NTOK / 256, 256, 0, stream>>>(tokens, NTOK);
  zero_i32_kernel<<<NPATCH / 256, 256, 0, stream>>>(lengths, NPATCH);
  scan_kernel<<<1, 32, 0, stream>>>(ent, seq, tokens, lengths);
  embed_patches_kernel<<<(NTOK * D_MODEL) / 256, 256, 0, stream>>>(tokens, byte_emb, x32, x16);

  // ---- 2 transformer encoder layers ----
  for (int l = 0; l < 2; ++l) {
    const f16* wqkv_l = wqkv_h + (size_t)l * 3 * D_MODEL * D_MODEL;
    const f16* wo_l   = wo_h   + (size_t)l * D_MODEL * D_MODEL;
    const f16* w1_l   = w1_h   + (size_t)l * FDIM * D_MODEL;
    const f16* w2_l   = w2_h   + (size_t)l * D_MODEL * FDIM;

    gemm_f16_kernel<0, true, false, false, true><<<gblk(NTOK, 3 * D_MODEL), 256, 0, stream>>>(
        x16, wqkv_l, bqkv + (size_t)l * 3 * D_MODEL, nullptr, nullptr, qkv32,
        NTOK, 3 * D_MODEL, D_MODEL);
    attention_kernel<<<NPATCH, 256, 0, stream>>>(qkv32, lengths, att16);
    gemm_f16_kernel<0, true, true, false, true><<<gblk(NTOK, D_MODEL), 256, 0, stream>>>(
        att16, wo_l, bo + (size_t)l * D_MODEL, x32, nullptr, y32,
        NTOK, D_MODEL, D_MODEL);
    ln_kernel<<<NTOK / 8, 256, 0, stream>>>(y32, ln1g + (size_t)l * D_MODEL,
                                            ln1b + (size_t)l * D_MODEL, x32, x16, NTOK);
    gemm_f16_kernel<1, true, false, true, false><<<gblk(NTOK, FDIM), 256, 0, stream>>>(
        x16, w1_l, b1 + (size_t)l * FDIM, nullptr, hf16, nullptr,
        NTOK, FDIM, D_MODEL);
    gemm_f16_kernel<0, true, true, false, true><<<gblk(NTOK, D_MODEL), 256, 0, stream>>>(
        hf16, w2_l, b2 + (size_t)l * D_MODEL, x32, nullptr, y32,
        NTOK, D_MODEL, FDIM);
    ln_kernel<<<NTOK / 8, 256, 0, stream>>>(y32, ln2g + (size_t)l * D_MODEL,
                                            ln2b + (size_t)l * D_MODEL, x32, x16, NTOK);
  }

  // ---- pool + outputs (emb f32 [4096,512], then lengths i32 [4096]) ----
  float* emb_out = (float*)d_out;
  int*   len_out = (int*)(emb_out + (size_t)NPATCH * D_MODEL);
  pool_kernel<<<NPATCH / 8, 256, 0, stream>>>(x32, lengths, emb_out, len_out);
}
